// FluxSingleStreamAttention_90993177133186
// MI455X (gfx1250) — matrix operand order repaired and checked by reading, not compiled, not run on hardware
//
#include <hip/hip_runtime.h>

#define DIM   3072
#define HEADS 24
#define HD    128
#define SEQ   4096
#define EPS   1e-6f
#define KSTEP 64

typedef __attribute__((ext_vector_type(16))) __bf16 v16bf;
typedef __attribute__((ext_vector_type(8)))  __bf16 v8bf;
typedef __attribute__((ext_vector_type(8)))  float  v8f;

// ---------------------------------------------------------------------------
// WMMA helper: D = A(16x32 bf16) * B(32x16 bf16) + C(16x16 f32)
// ---------------------------------------------------------------------------
static __device__ __forceinline__ v8f wmma_bf16(v16bf a, v16bf b, v8f c) {
  return __builtin_amdgcn_wmma_f32_16x16x32_bf16(
      /*neg_a=*/false, a, /*neg_b=*/false, b,
      /*c_mod=*/(short)0, c, /*reuse_a=*/false, /*reuse_b=*/false);
}

// A-matrix fragment (16x32, 16-bit): lane<16 holds K=0..15,
// lane>=16 holds K=16..31. Caller passes p = row_base + (lane>>4)*16.
static __device__ __forceinline__ v16bf frag_a(const __bf16* p) {
  return *(const v16bf*)p;
}

// B-matrix fragment (32x16, 16-bit): lane<16 = N, K contiguous per half-wave.
static __device__ __forceinline__ v16bf frag_b(const __bf16* p) {
  return *(const v16bf*)p;
}

// ---------------------------------------------------------------------------
// CDNA5 async copy: global -> LDS, 16B per lane, tracked by ASYNCcnt.
// ---------------------------------------------------------------------------
static __device__ __forceinline__ void async_copy_b128(void* lds,
                                                       const void* gptr) {
  unsigned lds_off = (unsigned)(unsigned long long)lds;
  asm volatile("global_load_async_to_lds_b128 %0, %1, off"
               :: "v"(lds_off), "v"(gptr)
               : "memory");
}
static __device__ __forceinline__ void async_wait0() {
  asm volatile("s_wait_asynccnt 0" ::: "memory");
}

// ---------------------------------------------------------------------------
// Kernel 1: f32 -> bf16 elementwise (hidden states)
// ---------------------------------------------------------------------------
__global__ void k_cvt_bf16(const float* __restrict__ in,
                           __bf16* __restrict__ out, int n) {
  int i = blockIdx.x * blockDim.x + threadIdx.x;
  int stride = gridDim.x * blockDim.x;
  for (; i < n; i += stride) out[i] = (__bf16)in[i];
}

// ---------------------------------------------------------------------------
// Kernel 2: W (K x N, f32, row-major) -> Wt (N x K, bf16) tiled transpose
// ---------------------------------------------------------------------------
__global__ __launch_bounds__(256)
void k_transpose_bf16(const float* __restrict__ W, __bf16* __restrict__ Wt,
                      int K, int N) {
  __shared__ __bf16 tile[64][68];
  int kb = blockIdx.x * 64;
  int nb = blockIdx.y * 64;
  int t = threadIdx.x;
#pragma unroll
  for (int i = 0; i < 16; ++i) {
    int idx = t + i * 256;
    int r = idx >> 6, c = idx & 63;
    tile[r][c] = (__bf16)W[(size_t)(kb + r) * N + nb + c];
  }
  __syncthreads();
#pragma unroll
  for (int i = 0; i < 16; ++i) {
    int idx = t + i * 256;
    int r = idx >> 6, c = idx & 63;               // r = n-local, c = k-local
    Wt[(size_t)(nb + r) * K + kb + c] = tile[c][r];
  }
}

// ---------------------------------------------------------------------------
// Kernel 3: fused projection GEMM + bias (+ RMSNorm + RoPE for Q/K)
//   X  : SEQ x DIM   (bf16, row-major)
//   Wt : 3072 x DIM  (bf16, N-major so B fragments are contiguous in K)
//   block tile = 64 rows x 128 cols (one head), 8 waves (4 x 2)
//   double-buffered K-step 64 with async global->LDS copies
//   mode 1: out[s][h*128+d] = rope(rmsnorm(xW+b))   (bf16)
//   mode 0: outVt[(h*128+d)*SEQ + s] = xW+b         (bf16, transposed for PV)
// ---------------------------------------------------------------------------
__global__ __launch_bounds__(256)
void k_qkv_gemm(const __bf16* __restrict__ X, const __bf16* __restrict__ Wt,
                const float* __restrict__ bias, const float* __restrict__ cosp,
                const float* __restrict__ sinp, const float* __restrict__ g,
                __bf16* __restrict__ outQK, __bf16* __restrict__ outVt,
                int mode) {
  // 48KB: double-buffered A(16K)+B(32K) tiles, overlapped with 32KB f32 stage
  __shared__ alignas(64) unsigned char smem[49152];
  __bf16 (*At)[64][KSTEP] = (__bf16 (*)[64][KSTEP])(smem);             // [2]
  __bf16 (*Bt)[128][KSTEP] = (__bf16 (*)[128][KSTEP])(smem + 16384);   // [2]
  float (*Sacc)[128] = (float (*)[128])(smem);                         // [64]

  const int t = threadIdx.x;
  const int lane = t & 31;
  const int wave = t >> 5;          // 0..7
  const int wm = wave & 3;          // row-wave 0..3 (16 rows each)
  const int wn = wave >> 2;         // col-wave 0..1 (64 cols each)
  const int M0 = blockIdx.x * 64;
  const int h = blockIdx.y;
  const int N0 = h * HD;
  const int m = lane & 15;
  const int kh = lane >> 4;

  const int ar = t >> 2, acA = (t & 3) * 16;     // A: 16 halves/thread
  const int bn = t >> 1, bcB = (t & 1) * 32;     // B: 32 halves/thread

  v8f acc[4];
#pragma unroll
  for (int j = 0; j < 4; ++j)
#pragma unroll
    for (int r = 0; r < 8; ++r) acc[j][r] = 0.0f;

  auto load_tiles = [&](int buf, int k0) {
    const __bf16* xs = &X[(size_t)(M0 + ar) * DIM + k0 + acA];
    async_copy_b128(&At[buf][ar][acA], xs);
    async_copy_b128(&At[buf][ar][acA + 8], xs + 8);
    const __bf16* wsrc = &Wt[(size_t)(N0 + bn) * DIM + k0 + bcB];
    async_copy_b128(&Bt[buf][bn][bcB], wsrc);
    async_copy_b128(&Bt[buf][bn][bcB + 8], wsrc + 8);
    async_copy_b128(&Bt[buf][bn][bcB + 16], wsrc + 16);
    async_copy_b128(&Bt[buf][bn][bcB + 24], wsrc + 24);
  };

  // prologue: fill buffer 0
  load_tiles(0, 0);
  async_wait0();
  __syncthreads();

  for (int k0 = 0; k0 < DIM; k0 += KSTEP) {
    const int buf = (k0 / KSTEP) & 1;
    if (k0 + KSTEP < DIM) load_tiles(buf ^ 1, k0 + KSTEP);

    // load phase: all fragments of this stage into distinct registers so the
    // ds_load_b128s batch and the WMMAs below issue back-to-back
    v16bf afr[2], bfr[2][4];
#pragma unroll
    for (int kc = 0; kc < 2; ++kc) {
      afr[kc] = frag_a(&At[buf][wm * 16 + m][kc * 32 + kh * 16]);
#pragma unroll
      for (int j = 0; j < 4; ++j)
        bfr[kc][j] =
            frag_b(&Bt[buf][wn * 64 + j * 16 + m][kc * 32 + kh * 16]);
    }
    // compute phase: 8 back-to-back WMMAs
#pragma unroll
    for (int kc = 0; kc < 2; ++kc)
#pragma unroll
      for (int j = 0; j < 4; ++j)
        acc[j] = wmma_bf16(afr[kc], bfr[kc][j], acc[j]);

    async_wait0();     // next stage's data has landed in LDS
    __syncthreads();   // all waves done reading current buffer
  }

  // stage f32 results for cross-wave epilogue (reuses tile LDS)
#pragma unroll
  for (int j = 0; j < 4; ++j)
#pragma unroll
    for (int r = 0; r < 8; ++r) {
      int row = wm * 16 + r + (kh << 3);
      int col = wn * 64 + j * 16 + m;
      Sacc[row][col] = acc[j][r];
    }
  __syncthreads();

  // epilogue: 256 threads, thread = (row 0..63) x (quarter 0..3 of head dim)
  const int row = t >> 2;
  const int dbase = (t & 3) * 32;
  const int s = M0 + row;
  float arr[32];
  float ss = 0.0f;
#pragma unroll
  for (int i = 0; i < 32; ++i) {
    float v = Sacc[row][dbase + i] + bias[N0 + dbase + i];
    arr[i] = v;
    ss += v * v;
  }

  if (mode == 1) {
    ss += __shfl_xor(ss, 1);
    ss += __shfl_xor(ss, 2);
    float rms = rsqrtf(ss * (1.0f / (float)HD) + EPS);
#pragma unroll
    for (int i = 0; i < 32; ++i) arr[i] = arr[i] * rms * g[dbase + i];
#pragma unroll
    for (int i = 0; i < 32; i += 2) {
      int d = dbase + i;
      float c0 = cosp[(size_t)s * HD + d];
      float s0 = sinp[(size_t)s * HD + d];
      float c1 = cosp[(size_t)s * HD + d + 1];
      float s1 = sinp[(size_t)s * HD + d + 1];
      float x0 = arr[i], x1 = arr[i + 1];
      outQK[(size_t)s * DIM + N0 + d]     = (__bf16)(x0 * c0 - x1 * s0);
      outQK[(size_t)s * DIM + N0 + d + 1] = (__bf16)(x1 * c1 + x0 * s1);
    }
  } else {
#pragma unroll
    for (int i = 0; i < 32; ++i)
      outVt[(size_t)(N0 + dbase + i) * SEQ + s] = (__bf16)arr[i];
  }
}

// ---------------------------------------------------------------------------
// Kernel 4: flash attention.
//   Q,K : SEQ x 3072 bf16 (post norm+rope);  Vt : (h*128+d) x SEQ bf16
//   block = (64 q rows, 1 head), 4 waves, each wave owns 16 q rows.
//   async global->LDS K/V tile streaming; online softmax; WMMA QK^T and PV.
// ---------------------------------------------------------------------------
__global__ __launch_bounds__(128)
void k_attn(const __bf16* __restrict__ Q, const __bf16* __restrict__ K,
            const __bf16* __restrict__ Vt, float* __restrict__ out) {
  __shared__ alignas(64) __bf16 Kt[64][HD];     // [token][d]   16 KB
  __shared__ alignas(64) __bf16 Vs[HD][64];     // [d][token]   16 KB
  __shared__ alignas(64) __bf16 Ps[4][16][64];  // per-wave P    8 KB

  const int t = threadIdx.x;
  const int lane = t & 31;
  const int w = t >> 5;             // wave 0..3
  const int h = blockIdx.y;
  const int s0 = blockIdx.x * 64 + w * 16;
  const int m = lane & 15;
  const int kh = lane >> 4;
  const float scale = 0.08838834764831845f;     // 1/sqrt(128)

  // Q fragments for this wave's 16 rows: 4 chunks of K-dim (head dim 128)
  v16bf qf[4];
#pragma unroll
  for (int c = 0; c < 4; ++c)
    qf[c] = frag_a(&Q[(size_t)(s0 + m) * DIM + h * HD + c * 32 + kh * 16]);

  v8f o[4];
  float mrun[8], lrun[8];
#pragma unroll
  for (int j = 0; j < 4; ++j)
#pragma unroll
    for (int r = 0; r < 8; ++r) o[j][r] = 0.0f;
#pragma unroll
  for (int r = 0; r < 8; ++r) { mrun[r] = -1e30f; lrun[r] = 0.0f; }

  for (int kb = 0; kb < SEQ; kb += 64) {
    // cooperative async tile loads (128 threads; 16B granules direct to LDS)
#pragma unroll
    for (int i = 0; i < 4; ++i) {
      int c = t + i * 128;
      int tok = c >> 3, seg = (c & 7) * 16;
      const __bf16* src = &K[(size_t)(kb + tok) * DIM + h * HD + seg];
      async_copy_b128(&Kt[tok][seg], src);
      async_copy_b128(&Kt[tok][seg + 8], src + 8);
    }
#pragma unroll
    for (int i = 0; i < 4; ++i) {
      int c = t + i * 128;
      int d = c >> 2, seg = (c & 3) * 16;
      const __bf16* src = &Vt[(size_t)(h * HD + d) * SEQ + kb + seg];
      async_copy_b128(&Vs[d][seg], src);
      async_copy_b128(&Vs[d][seg + 8], src + 8);
    }
    async_wait0();
    __syncthreads();

    // S = scale * Q K^T : 16x64 per wave (4 N-tiles x 4 K-chunks)
    v8f sacc[4];
#pragma unroll
    for (int j = 0; j < 4; ++j)
#pragma unroll
      for (int r = 0; r < 8; ++r) sacc[j][r] = 0.0f;
#pragma unroll
    for (int j = 0; j < 4; ++j) {
      // batch the 4 K-fragment loads, then 4 back-to-back WMMAs
      v16bf kf[4];
#pragma unroll
      for (int c = 0; c < 4; ++c)
        kf[c] = frag_b(&Kt[j * 16 + m][c * 32 + kh * 16]);
#pragma unroll
      for (int c = 0; c < 4; ++c)
        sacc[j] = wmma_bf16(qf[c], kf[c], sacc[j]);
    }

    // online softmax: per-row stats (row r lives in VGPR r of each half-wave)
    float bmax[8];
#pragma unroll
    for (int r = 0; r < 8; ++r) bmax[r] = -1e30f;
#pragma unroll
    for (int j = 0; j < 4; ++j)
#pragma unroll
      for (int r = 0; r < 8; ++r) {
        float v = sacc[j][r] * scale;
        sacc[j][r] = v;
        bmax[r] = fmaxf(bmax[r], v);
      }
#pragma unroll
    for (int r = 0; r < 8; ++r) {
      bmax[r] = fmaxf(bmax[r], __shfl_xor(bmax[r], 1));
      bmax[r] = fmaxf(bmax[r], __shfl_xor(bmax[r], 2));
      bmax[r] = fmaxf(bmax[r], __shfl_xor(bmax[r], 4));
      bmax[r] = fmaxf(bmax[r], __shfl_xor(bmax[r], 8));
    }
    float alpha[8], rsum[8];
#pragma unroll
    for (int r = 0; r < 8; ++r) {
      float mn = fmaxf(mrun[r], bmax[r]);
      alpha[r] = __expf(mrun[r] - mn);
      mrun[r] = mn;
      rsum[r] = 0.0f;
    }

    // P = exp(S - m_new), stage bf16 to LDS in A-matrix memory order
#pragma unroll
    for (int j = 0; j < 4; ++j)
#pragma unroll
      for (int r = 0; r < 8; ++r) {
        float p = __expf(sacc[j][r] - mrun[r]);
        rsum[r] += p;
        Ps[w][r + (kh << 3)][j * 16 + m] = (__bf16)p;
      }
#pragma unroll
    for (int r = 0; r < 8; ++r) {
      rsum[r] += __shfl_xor(rsum[r], 1);
      rsum[r] += __shfl_xor(rsum[r], 2);
      rsum[r] += __shfl_xor(rsum[r], 4);
      rsum[r] += __shfl_xor(rsum[r], 8);
      lrun[r] = lrun[r] * alpha[r] + rsum[r];
    }

    // O = O * alpha + P @ V  (LDS P is wave-private; DS ops in-order per wave)
#pragma unroll
    for (int jd = 0; jd < 4; ++jd)
#pragma unroll
      for (int r = 0; r < 8; ++r) o[jd][r] *= alpha[r];
#pragma unroll
    for (int c2 = 0; c2 < 2; ++c2) {
      // batch P fragment + 4 V fragments, then 4 back-to-back WMMAs
      v16bf pf = frag_a(&Ps[w][m][c2 * 32 + kh * 16]);
      v16bf vf[4];
#pragma unroll
      for (int jd = 0; jd < 4; ++jd)
        vf[jd] = frag_b(&Vs[jd * 16 + m][c2 * 32 + kh * 16]);
#pragma unroll
      for (int jd = 0; jd < 4; ++jd)
        o[jd] = wmma_bf16(pf, vf[jd], o[jd]);
    }
    __syncthreads();   // before next iteration overwrites Kt / Vs
  }

  // normalize and store f32 output: out[s][h*128 + d]
#pragma unroll
  for (int r = 0; r < 8; ++r) {
    float inv = 1.0f / lrun[r];
    int s = s0 + r + (kh << 3);
#pragma unroll
    for (int jd = 0; jd < 4; ++jd)
      out[(size_t)s * DIM + h * HD + jd * 16 + m] = o[jd][r] * inv;
  }
}

// ---------------------------------------------------------------------------
// Host launcher
// ---------------------------------------------------------------------------
extern "C" void kernel_launch(void* const* d_in, const int* in_sizes, int n_in,
                              void* d_out, int out_size, void* d_ws,
                              size_t ws_size, hipStream_t stream) {
  const float* hs   = (const float*)d_in[0];
  const float* cosp = (const float*)d_in[1];
  const float* sinp = (const float*)d_in[2];
  const float* Wq   = (const float*)d_in[3];
  const float* bq   = (const float*)d_in[4];
  const float* Wk   = (const float*)d_in[5];
  const float* bk   = (const float*)d_in[6];
  const float* Wv   = (const float*)d_in[7];
  const float* bv   = (const float*)d_in[8];
  const float* gq   = (const float*)d_in[9];
  const float* gk   = (const float*)d_in[10];

  char* ws = (char*)d_ws;
  size_t off = 0;
  __bf16* Xh  = (__bf16*)(ws + off); off += (size_t)SEQ * DIM * 2;
  __bf16* Wqt = (__bf16*)(ws + off); off += (size_t)DIM * DIM * 2;
  __bf16* Wkt = (__bf16*)(ws + off); off += (size_t)DIM * DIM * 2;
  __bf16* Wvt = (__bf16*)(ws + off); off += (size_t)DIM * DIM * 2;
  __bf16* Qh  = (__bf16*)(ws + off); off += (size_t)SEQ * DIM * 2;
  __bf16* Kh  = (__bf16*)(ws + off); off += (size_t)SEQ * DIM * 2;
  __bf16* Vth = (__bf16*)(ws + off); off += (size_t)SEQ * DIM * 2;  // 150 MB

  // 1. convert hidden states to bf16
  k_cvt_bf16<<<2048, 256, 0, stream>>>(hs, Xh, SEQ * DIM);

  // 2. transpose + convert weights (N-major bf16)
  dim3 tg(DIM / 64, DIM / 64);
  k_transpose_bf16<<<tg, 256, 0, stream>>>(Wq, Wqt, DIM, DIM);
  k_transpose_bf16<<<tg, 256, 0, stream>>>(Wk, Wkt, DIM, DIM);
  k_transpose_bf16<<<tg, 256, 0, stream>>>(Wv, Wvt, DIM, DIM);

  // 3. projections (Q/K fused with RMSNorm+RoPE; V transposed for PV WMMA)
  dim3 gg(SEQ / 64, HEADS);
  k_qkv_gemm<<<gg, 256, 0, stream>>>(Xh, Wqt, bq, cosp, sinp, gq, Qh, nullptr, 1);
  k_qkv_gemm<<<gg, 256, 0, stream>>>(Xh, Wkt, bk, cosp, sinp, gk, Kh, nullptr, 1);
  k_qkv_gemm<<<gg, 256, 0, stream>>>(Xh, Wvt, bv, cosp, sinp, gq, nullptr, Vth, 0);

  // 4. attention
  k_attn<<<gg, 128, 0, stream>>>(Qh, Kh, Vth, (float*)d_out);
}
